// EgnnQNet_38448547234262
// MI455X (gfx1250) — matrix operand order repaired and checked by reading, not compiled, not run on hardware
//
#include <hip/hip_runtime.h>

// ---------------------------------------------------------------------------
// Fused EGNN Q-network for MI455X (gfx1250, wave32, WMMA f16 16x16x32).
// 4 batch items per workgroup (256 thr = 8 waves): 24 edge tiles -> 3 per
// wave (perfect balance); node phase -> 8 waves (4 items x {h,v} branches).
// Weights pre-packed to f16 WMMA B-fragment layout in d_ws by pack kernels.
// ---------------------------------------------------------------------------

typedef _Float16 half_t;
typedef __attribute__((ext_vector_type(16))) _Float16 v16h;
typedef __attribute__((ext_vector_type(8)))  _Float16 v8h;
typedef __attribute__((ext_vector_type(8)))  float    v8f;

#define HID       128
#define N_NODES   10
#define EDGES     90
#define ETILES    6
#define ITEMS     4                  // batch items per workgroup
#define TILES_TOT (ITEMS * ETILES)   // 24 edge tiles per workgroup
#define BATCH     4096
#define ROWS      (16 * HID)         // one 16-row tile of features

// ---------------- device helpers ----------------

__device__ __forceinline__ float silu_f(float x) { return x / (1.f + __expf(-x)); }

__device__ __forceinline__ v8f wmma_f16(v16h a, v16h b, v8f c) {
  // D = A(16x32 f16) * B(32x16 f16) + C(16x16 f32)
  return __builtin_amdgcn_wmma_f32_16x16x32_f16(false, a, false, b, (short)0, c,
                                                false, false);
}

// A fragment (16x32 f16) built from an f32 row in LDS.
// Layout (ISA 7.12.2, 16-bit A 16x32): lane<16 -> K {0..7,16..23}; lane>=16 -> +8.
__device__ __forceinline__ v16h a_from_f32(const float* rowptr, int fb, int lane) {
  const int off = (lane & 16) ? 8 : 0;
  const float* p0 = rowptr + fb + off;
  const float* p1 = rowptr + fb + 16 + off;
  v16h a;
#pragma unroll
  for (int i = 0; i < 8; ++i) {
    a[i]     = (half_t)p0[i];
    a[8 + i] = (half_t)p1[i];
  }
  return a;
}

// Same, but source row already f16 in LDS (two 16-byte vector loads).
__device__ __forceinline__ v16h a_from_f16(const half_t* rowptr, int fb, int lane) {
  const int off = (lane & 16) ? 8 : 0;
  v8h lo = *(const v8h*)(rowptr + fb + off);
  v8h hi = *(const v8h*)(rowptr + fb + 16 + off);
  v16h a;
#pragma unroll
  for (int i = 0; i < 8; ++i) {
    a[i]     = lo[i];
    a[8 + i] = hi[i];
  }
  return a;
}

// B fragment load from packed global layout (one 32B vector load per lane).
__device__ __forceinline__ v16h b_frag(const half_t* pk, int kc, int nt, int lane) {
  return *(const v16h*)(pk + ((((kc * 8 + nt) * 32) + lane) << 4));
}

// Butterfly sum over the 16 lanes of each wave half (lanes 0-15 / 16-31).
__device__ __forceinline__ void reduce_half16(float p[8]) {
#pragma unroll
  for (int m = 1; m <= 8; m <<= 1) {
#pragma unroll
    for (int g = 0; g < 8; ++g) p[g] += __shfl_xor(p[g], m, 32);
  }
}

// ---------------- weight packing ----------------
// Pack f32 weight (fanIn x 128, row-major) into f16 B-fragment layout:
// dst[((kc*8+nt)*32+lane)*16+hi] = W[kc*32 + (hi&7) + ((hi&8)?16:0) + (lane>=16?8:0)][nt*16+(lane&15)]
__global__ void pack_b_kernel(const float* __restrict__ w, half_t* __restrict__ dst,
                              int fanIn, int total) {
  int idx = blockIdx.x * blockDim.x + threadIdx.x;
  if (idx >= total) return;
  int hi   = idx & 15;
  int lane = (idx >> 4) & 31;
  int nt   = (idx >> 9) & 7;
  int kc   = idx >> 12;
  int K = kc * 32 + (hi & 7) + ((hi & 8) << 1) + ((lane & 16) ? 8 : 0);
  int n = nt * 16 + (lane & 15);
  dst[idx] = (K < fanIn) ? (half_t)w[K * HID + n] : (half_t)0.f;
}

// ---------------- kernel arguments ----------------

struct EgnnArgs {
  const float* obs;      // (BATCH, 200)
  const float* actions;  // (N, 2)
  float* out;            // (BATCH)
  const half_t* pk_embin;  const float* b_embin;
  const half_t* pk_embout; const float* b_embout;
  const float* fc_w; const float* fc_b;
  const half_t* pk_e0[3]; const float* e0w[3]; const float* e0b[3];
  const half_t* pk_e1[3]; const float* e1b[3];
  const half_t* pk_c0[3]; const float* c0b[3];
  const float* c1w[3]; const float* c1b[3];
  const half_t* pk_n0[3]; const float* n0b[3];
  const half_t* pk_n1[3]; const float* n1b[3];
  const half_t* pk_v0[3]; const float* v0b[3];
  const float* v1w[3]; const float* v1b[3];
};

// ---------------- fused EGNN kernel ----------------

__launch_bounds__(256, 1)
__global__ void egnn_fused_kernel(EgnnArgs A) {
  __shared__ float  h_s[ITEMS * ROWS];        // node features (rows>=10 pad)
  __shared__ float  aggh[ITEMS * ROWS];       // segment_sum(m)
  __shared__ float  delta[ITEMS * ROWS];      // n-branch residual
  __shared__ half_t scrW[8][ROWS];            // per-wave edge f16 scratch
  __shared__ half_t scrN[ITEMS][ROWS];        // node-branch f16 scratch
  __shared__ float  xs[ITEMS][16][2], vs[ITEMS][16][2], aggc[ITEMS][16][2];
  __shared__ float  ea[TILES_TOT][16], rad[TILES_TOT][16];
  __shared__ float  dnx[TILES_TOT][16], dny[TILES_TOT][16];

  const int tid  = threadIdx.x;
  const int lane = tid & 31;
  const int wave = tid >> 5;
  const int lane15 = lane & 15;
  const int half   = lane >> 4;
  const int b0 = blockIdx.x * ITEMS;

  // ---- init coordinates / velocities (4 items x 10 nodes) ----
  if (tid < ITEMS * N_NODES) {
    int item = tid / N_NODES, n = tid % N_NODES;
    const float* o = A.obs + (size_t)(b0 + item) * (N_NODES * 20);
    xs[item][n][0] = o[n * 20 + 16];
    xs[item][n][1] = o[n * 20 + 17];
    vs[item][n][0] = A.actions[((b0 + item) * N_NODES + n) * 2 + 0];
    vs[item][n][1] = A.actions[((b0 + item) * N_NODES + n) * 2 + 1];
  }

  // ---- h = emb_in(inv): waves 0..3, one item each (K=16 in one 32-chunk) ----
  if (wave < ITEMS) {
    const int item = wave;
    const float* o = A.obs + (size_t)(b0 + item) * (N_NODES * 20);
    float* hh = h_s + item * ROWS;
    v8f acc[8];
#pragma unroll
    for (int nt = 0; nt < 8; ++nt) {
      float bv = A.b_embin[nt * 16 + lane15];
#pragma unroll
      for (int g = 0; g < 8; ++g) acc[nt][g] = bv;
    }
    v16h a;
#pragma unroll
    for (int hi2 = 0; hi2 < 16; ++hi2) {
      int kl = (hi2 & 7) + ((hi2 & 8) << 1) + ((lane & 16) ? 8 : 0);
      float v = (kl < 16 && lane15 < N_NODES) ? o[lane15 * 20 + kl] : 0.f;
      a[hi2] = (half_t)v;
    }
#pragma unroll
    for (int nt = 0; nt < 8; ++nt)
      acc[nt] = wmma_f16(a, b_frag(A.pk_embin, 0, nt, lane), acc[nt]);
#pragma unroll
    for (int nt = 0; nt < 8; ++nt) {
#pragma unroll
      for (int g = 0; g < 8; ++g)
        hh[(g + half * 8) * HID + nt * 16 + lane15] = acc[nt][g];
    }
  }
  __syncthreads();

  // =============== message-passing layers ===============
  for (int l = 0; l < 3; ++l) {
    // zero aggregates
    for (int i = tid; i < ITEMS * ROWS; i += 256) aggh[i] = 0.f;
    if (tid < ITEMS * 32) ((float*)aggc)[tid] = 0.f;
    __syncthreads();

    // -------- edge phase: 24 tiles over 8 waves, 3 rounds each, balanced ----
    for (int rt = 0; rt < 3; ++rt) {
      const int t    = wave * 3 + rt;       // global tile 0..23
      const int item = t / ETILES;
      const int tile = t % ETILES;
      float* hh  = h_s  + item * ROWS;
      float* agh = aggh + item * ROWS;
      half_t* scr = scrW[wave];

      // geometry for this tile (lanes 0-15, one edge each)
      if (lane < 16) {
        int m = lane, e = tile * 16 + m;
        float dx = 0.f, dy = 0.f, rv = 0.f;
        if (e < EDGES) {
          int rr = e / 9, cc = e % 9; cc += (cc >= rr);
          dx = xs[item][rr][0] - xs[item][cc][0];
          dy = xs[item][rr][1] - xs[item][cc][1];
          rv = dx * dx + dy * dy;
        }
        rad[t][m] = rv;
        float inv = 1.f / (sqrtf(rv) + 1.f);
        dnx[t][m] = dx * inv;
        dny[t][m] = dy * inv;
        if (l == 0) ea[t][m] = rv;   // edge_attr = initial radial
      }

      // per-lane A-row edge (rows of the 16x* A tiles)
      int eA = tile * 16 + lane15;
      int eC = eA < EDGES ? eA : 0;
      int rr = eC / 9, cc = eC % 9; cc += (cc >= rr);

      v8f acc[8];
      const float* e0w = A.e0w[l];
      const float* e0b = A.e0b[l];
      // C init: bias + rank-1 contributions of features 256 (radial), 257 (ea)
#pragma unroll
      for (int nt = 0; nt < 8; ++nt) {
        int n = nt * 16 + lane15;
        float bv   = e0b[n];
        float w256 = e0w[256 * HID + n];
        float w257 = e0w[257 * HID + n];
#pragma unroll
        for (int g = 0; g < 8; ++g) {
          int m = g + half * 8;
          acc[nt][g] = bv + rad[t][m] * w256 + ea[t][m] * w257;
        }
      }
      // e0: K-chunks 0-3 from h[rows], 4-7 from h[cols]
      {
        const half_t* pk = A.pk_e0[l];
#pragma unroll
        for (int kc = 0; kc < 8; ++kc) {
          const float* rowp = hh + (kc < 4 ? rr : cc) * HID;
          v16h a = a_from_f32(rowp, (kc & 3) * 32, lane);
#pragma unroll
          for (int nt = 0; nt < 8; ++nt)
            acc[nt] = wmma_f16(a, b_frag(pk, kc, nt, lane), acc[nt]);
        }
      }
#pragma unroll
      for (int nt = 0; nt < 8; ++nt) {
        int n = nt * 16 + lane15;
#pragma unroll
        for (int g = 0; g < 8; ++g)
          scr[(g + half * 8) * HID + n] = (half_t)silu_f(acc[nt][g]);
      }

      // e1: m = silu(e1 @ t0)
      {
        const float* e1b = A.e1b[l];
#pragma unroll
        for (int nt = 0; nt < 8; ++nt) {
          float bv = e1b[nt * 16 + lane15];
#pragma unroll
          for (int g = 0; g < 8; ++g) acc[nt][g] = bv;
        }
        const half_t* pk = A.pk_e1[l];
#pragma unroll
        for (int kc = 0; kc < 4; ++kc) {
          v16h a = a_from_f16(scr + lane15 * HID, kc * 32, lane);
#pragma unroll
          for (int nt = 0; nt < 8; ++nt)
            acc[nt] = wmma_f16(a, b_frag(pk, kc, nt, lane), acc[nt]);
        }
      }
      // silu -> store m to scratch (for c0) + segment-sum into agg_h
#pragma unroll
      for (int nt = 0; nt < 8; ++nt) {
        int n = nt * 16 + lane15;
#pragma unroll
        for (int g = 0; g < 8; ++g) {
          int m = g + half * 8;
          float s = silu_f(acc[nt][g]);
          scr[m * HID + n] = (half_t)s;
          int e = tile * 16 + m;
          if (e < EDGES) atomicAdd(&agh[(e / 9) * HID + n], s);
        }
      }

      // c0 / c1: per-edge scalar gate for coordinate update
      {
        const float* c0b = A.c0b[l];
#pragma unroll
        for (int nt = 0; nt < 8; ++nt) {
          float bv = c0b[nt * 16 + lane15];
#pragma unroll
          for (int g = 0; g < 8; ++g) acc[nt][g] = bv;
        }
        const half_t* pk = A.pk_c0[l];
#pragma unroll
        for (int kc = 0; kc < 4; ++kc) {
          v16h a = a_from_f16(scr + lane15 * HID, kc * 32, lane);
#pragma unroll
          for (int nt = 0; nt < 8; ++nt)
            acc[nt] = wmma_f16(a, b_frag(pk, kc, nt, lane), acc[nt]);
        }
        const float* c1w = A.c1w[l];
        float p[8];
#pragma unroll
        for (int g = 0; g < 8; ++g) p[g] = 0.f;
#pragma unroll
        for (int nt = 0; nt < 8; ++nt) {
          float wv = c1w[nt * 16 + lane15];
#pragma unroll
          for (int g = 0; g < 8; ++g) p[g] += wv * silu_f(acc[nt][g]);
        }
        reduce_half16(p);
        float c1b = A.c1b[l][0];
        if (lane15 == 0) {
#pragma unroll
          for (int g = 0; g < 8; ++g) {
            int m = g + half * 8, e = tile * 16 + m;
            if (e < EDGES) {
              float s = (p[g] + c1b) * (1.f / 9.f);  // trans / DEG
              int r2 = e / 9;
              atomicAdd(&aggc[item][r2][0], dnx[t][m] * s);
              atomicAdd(&aggc[item][r2][1], dny[t][m] * s);
            }
          }
        }
      }
    }
    __syncthreads();

    // -------- node phase: 8 waves = 4 items x {h-branch, v-branch} ----------
    {
      const int item = wave >> 1;
      float* hh  = h_s  + item * ROWS;
      float* agh = aggh + item * ROWS;
      if ((wave & 1) == 0) {
        // h-branch: delta = n1 @ silu(n0 @ [h | agg_h])
        v8f acc[8];
        const float* n0b = A.n0b[l];
#pragma unroll
        for (int nt = 0; nt < 8; ++nt) {
          float bv = n0b[nt * 16 + lane15];
#pragma unroll
          for (int g = 0; g < 8; ++g) acc[nt][g] = bv;
        }
        {
          const half_t* pk = A.pk_n0[l];
#pragma unroll
          for (int kc = 0; kc < 8; ++kc) {
            const float* rowp = (kc < 4 ? hh : agh) + lane15 * HID;
            v16h a = a_from_f32(rowp, (kc & 3) * 32, lane);
#pragma unroll
            for (int nt = 0; nt < 8; ++nt)
              acc[nt] = wmma_f16(a, b_frag(pk, kc, nt, lane), acc[nt]);
          }
        }
        half_t* sn = scrN[item];
#pragma unroll
        for (int nt = 0; nt < 8; ++nt) {
#pragma unroll
          for (int g = 0; g < 8; ++g)
            sn[(g + half * 8) * HID + nt * 16 + lane15] =
                (half_t)silu_f(acc[nt][g]);
        }
        const float* n1b = A.n1b[l];
#pragma unroll
        for (int nt = 0; nt < 8; ++nt) {
          float bv = n1b[nt * 16 + lane15];
#pragma unroll
          for (int g = 0; g < 8; ++g) acc[nt][g] = bv;
        }
        {
          const half_t* pk = A.pk_n1[l];
#pragma unroll
          for (int kc = 0; kc < 4; ++kc) {
            v16h a = a_from_f16(sn + lane15 * HID, kc * 32, lane);
#pragma unroll
            for (int nt = 0; nt < 8; ++nt)
              acc[nt] = wmma_f16(a, b_frag(pk, kc, nt, lane), acc[nt]);
          }
        }
        float* dl = delta + item * ROWS;
#pragma unroll
        for (int nt = 0; nt < 8; ++nt) {
#pragma unroll
          for (int g = 0; g < 8; ++g)
            dl[(g + half * 8) * HID + nt * 16 + lane15] = acc[nt][g];
        }
      } else {
        // v-branch: v = (v1 . silu(v0 @ h)) * v + agg_c ; x += v
        v8f acc[8];
        const float* v0b = A.v0b[l];
#pragma unroll
        for (int nt = 0; nt < 8; ++nt) {
          float bv = v0b[nt * 16 + lane15];
#pragma unroll
          for (int g = 0; g < 8; ++g) acc[nt][g] = bv;
        }
        {
          const half_t* pk = A.pk_v0[l];
#pragma unroll
          for (int kc = 0; kc < 4; ++kc) {
            v16h a = a_from_f32(hh + lane15 * HID, kc * 32, lane);
#pragma unroll
            for (int nt = 0; nt < 8; ++nt)
              acc[nt] = wmma_f16(a, b_frag(pk, kc, nt, lane), acc[nt]);
          }
        }
        const float* v1w = A.v1w[l];
        float p[8];
#pragma unroll
        for (int g = 0; g < 8; ++g) p[g] = 0.f;
#pragma unroll
        for (int nt = 0; nt < 8; ++nt) {
          float wv = v1w[nt * 16 + lane15];
#pragma unroll
          for (int g = 0; g < 8; ++g) p[g] += wv * silu_f(acc[nt][g]);
        }
        reduce_half16(p);
        float v1b = A.v1b[l][0];
        if (lane15 == 0) {
#pragma unroll
          for (int g = 0; g < 8; ++g) {
            int m = g + half * 8;
            if (m < N_NODES) {
              float sc = p[g] + v1b;
              float vn0 = sc * vs[item][m][0] + aggc[item][m][0];
              float vn1 = sc * vs[item][m][1] + aggc[item][m][1];
              vs[item][m][0] = vn0; vs[item][m][1] = vn1;
              xs[item][m][0] += vn0; xs[item][m][1] += vn1;
            }
          }
        }
      }
    }
    __syncthreads();
    // residual: h += delta
    for (int i = tid; i < ITEMS * ROWS; i += 256) h_s[i] += delta[i];
    __syncthreads();
  }

  // ====== head: q = tanh(emb_out(h)) @ fc ; mean over nodes (waves 0..3) ====
  if (wave < ITEMS) {
    const int item = wave;
    const float* hh = h_s + item * ROWS;
    v8f acc[8];
#pragma unroll
    for (int nt = 0; nt < 8; ++nt) {
      float bv = A.b_embout[nt * 16 + lane15];
#pragma unroll
      for (int g = 0; g < 8; ++g) acc[nt][g] = bv;
    }
#pragma unroll
    for (int kc = 0; kc < 4; ++kc) {
      v16h a = a_from_f32(hh + lane15 * HID, kc * 32, lane);
#pragma unroll
      for (int nt = 0; nt < 8; ++nt)
        acc[nt] = wmma_f16(a, b_frag(A.pk_embout, kc, nt, lane), acc[nt]);
    }
    const float* fcw = A.fc_w;
    float fcb = A.fc_b[0];
    float p[8];
#pragma unroll
    for (int g = 0; g < 8; ++g) p[g] = 0.f;
#pragma unroll
    for (int nt = 0; nt < 8; ++nt) {
      float wv = fcw[nt * 16 + lane15];
#pragma unroll
      for (int g = 0; g < 8; ++g) p[g] += wv * tanhf(acc[nt][g]);
    }
    reduce_half16(p);
    float qm = 0.f;
#pragma unroll
    for (int g = 0; g < 8; ++g) {
      int m = g + half * 8;
      if (m < N_NODES) qm += p[g] + fcb;
    }
    float tot = qm + __shfl_xor(qm, 16, 32);
    if (lane == 0) A.out[b0 + item] = tot * (1.f / (float)N_NODES);
  }
}

// ---------------- host launcher ----------------

extern "C" void kernel_launch(void* const* d_in, const int* in_sizes, int n_in,
                              void* d_out, int out_size, void* d_ws, size_t ws_size,
                              hipStream_t stream) {
  (void)in_sizes; (void)n_in; (void)out_size; (void)ws_size;
  // Input order: cent_obs, actions, then params pytree leaves in sorted-key
  // order (emb_in, emb_out, fc, l0, l1, l2; each linear as {b, w}; each layer
  // dict as c0,c1,e0,e1,n0,n1,v0,v1), then rows, cols.
  int i = 0;
  const float* obs      = (const float*)d_in[i++];
  const float* actions  = (const float*)d_in[i++];
  const float* embin_b  = (const float*)d_in[i++];
  const float* embin_w  = (const float*)d_in[i++];
  const float* embout_b = (const float*)d_in[i++];
  const float* embout_w = (const float*)d_in[i++];
  const float* fc_b     = (const float*)d_in[i++];
  const float* fc_w     = (const float*)d_in[i++];
  const float *c0b[3], *c0w[3], *c1b[3], *c1w[3], *e0b[3], *e0w[3], *e1b[3],
      *e1w[3], *n0b[3], *n0w[3], *n1b[3], *n1w[3], *v0b[3], *v0w[3], *v1b[3],
      *v1w[3];
  for (int l = 0; l < 3; ++l) {
    c0b[l] = (const float*)d_in[i++]; c0w[l] = (const float*)d_in[i++];
    c1b[l] = (const float*)d_in[i++]; c1w[l] = (const float*)d_in[i++];
    e0b[l] = (const float*)d_in[i++]; e0w[l] = (const float*)d_in[i++];
    e1b[l] = (const float*)d_in[i++]; e1w[l] = (const float*)d_in[i++];
    n0b[l] = (const float*)d_in[i++]; n0w[l] = (const float*)d_in[i++];
    n1b[l] = (const float*)d_in[i++]; n1w[l] = (const float*)d_in[i++];
    v0b[l] = (const float*)d_in[i++]; v0w[l] = (const float*)d_in[i++];
    v1b[l] = (const float*)d_in[i++]; v1w[l] = (const float*)d_in[i++];
  }
  // rows/cols (d_in[i], d_in[i+1]) are an analytic pattern; unused.

  half_t* ws = (half_t*)d_ws;
  size_t off = 0;
  auto pack = [&](const float* w, int fanIn, int chunks) -> const half_t* {
    half_t* dst = ws + off;
    int total = chunks * 4096;  // chunks * 8 ntiles * 32 lanes * 16 halves
    pack_b_kernel<<<(total + 255) / 256, 256, 0, stream>>>(w, dst, fanIn, total);
    off += (size_t)total;
    return dst;
  };

  EgnnArgs A;
  A.obs = obs; A.actions = actions; A.out = (float*)d_out;
  A.b_embin = embin_b; A.b_embout = embout_b; A.fc_w = fc_w; A.fc_b = fc_b;
  A.pk_embin  = pack(embin_w, 16, 1);
  A.pk_embout = pack(embout_w, 128, 4);
  for (int l = 0; l < 3; ++l) {
    A.pk_e0[l] = pack(e0w[l], 258, 8);   // K 0..255 via WMMA; rows 256/257 rank-1
    A.pk_e1[l] = pack(e1w[l], 128, 4);
    A.pk_c0[l] = pack(c0w[l], 128, 4);
    A.pk_n0[l] = pack(n0w[l], 256, 8);
    A.pk_n1[l] = pack(n1w[l], 128, 4);
    A.pk_v0[l] = pack(v0w[l], 128, 4);
    A.e0w[l] = e0w[l]; A.e0b[l] = e0b[l]; A.e1b[l] = e1b[l];
    A.c0b[l] = c0b[l]; A.c1w[l] = c1w[l]; A.c1b[l] = c1b[l];
    A.n0b[l] = n0b[l]; A.n1b[l] = n1b[l];
    A.v0b[l] = v0b[l]; A.v1w[l] = v1w[l]; A.v1b[l] = v1b[l];
  }

  egnn_fused_kernel<<<BATCH / ITEMS, 256, 0, stream>>>(A);
}
